// PharmaSAE_3839700763074
// MI455X (gfx1250) — matrix operand order repaired and checked by reading, not compiled, not run on hardware
//
#include <hip/hip_runtime.h>
#include <hip/hip_bf16.h>

typedef __bf16 bf16_t;
typedef bf16_t v8bf  __attribute__((ext_vector_type(8)));
typedef bf16_t v16bf __attribute__((ext_vector_type(16)));
typedef float  v8f   __attribute__((ext_vector_type(8)));
typedef int    v4i_vs __attribute__((vector_size(16)));   // matches builtin param type

#define BB 16384
#define DD 1024
#define FF 8192
#define KK 30
#define LDSS 40   // padded LDS k-stride (bf16 elems): 80B rows, 16B aligned

#define HAS_ASYNC_LDS __has_builtin(__builtin_amdgcn_global_load_async_to_lds_b128)
#define AS_GLOBAL __attribute__((address_space(1)))
#define AS_SHARED __attribute__((address_space(3)))

// 16B global -> LDS copy: async (ASYNCcnt-tracked) on CDNA5, sync fallback otherwise
__device__ __forceinline__ void cp16(const bf16_t* g, bf16_t* l)
{
#if HAS_ASYNC_LDS
    __builtin_amdgcn_global_load_async_to_lds_b128(
        (AS_GLOBAL v4i_vs*)g, (AS_SHARED v4i_vs*)l, 0, 0);
#else
    *(v8bf*)l = *(const v8bf*)g;
#endif
}

template <int N>
__device__ __forceinline__ void wait_async()
{
#if HAS_ASYNC_LDS
#if __has_builtin(__builtin_amdgcn_s_wait_asynccnt)
    __builtin_amdgcn_s_wait_asynccnt((short)N);
#else
    asm volatile("s_wait_asynccnt %0" ::"n"(N) : "memory");
#endif
#endif
}

// ---------------- prep kernels ----------------
__global__ void k_center_bf16(const float* __restrict__ x,
                              const float* __restrict__ dec_b,
                              bf16_t* __restrict__ xc)
{
    int i = (blockIdx.x * 256 + threadIdx.x) * 8;
    int d = i & (DD - 1);
    v8bf r;
#pragma unroll
    for (int j = 0; j < 8; ++j)
        r[j] = (bf16_t)(x[i + j] - dec_b[d + j]);
    *(v8bf*)(xc + i) = r;
}

__global__ void k_cvt_bf16(const float* __restrict__ w, bf16_t* __restrict__ o)
{
    int i = (blockIdx.x * 256 + threadIdx.x) * 8;
    v8bf r;
#pragma unroll
    for (int j = 0; j < 8; ++j)
        r[j] = (bf16_t)w[i + j];
    *(v8bf*)(o + i) = r;
}

// tiled transpose dec_w [DD][FF] -> dec_wT [FF][DD], both sides coalesced
__global__ void __launch_bounds__(256)
k_transpose(const float* __restrict__ dw, float* __restrict__ dwt)
{
    __shared__ float t[32][33];
    const int f0 = blockIdx.x * 32;
    const int d0 = blockIdx.y * 32;
    const int tx = threadIdx.x & 31;
    const int ty = threadIdx.x >> 5;  // 0..7
#pragma unroll
    for (int r = 0; r < 4; ++r)
        t[ty + 8 * r][tx] = dw[(size_t)(d0 + ty + 8 * r) * FF + f0 + tx];
    __syncthreads();
#pragma unroll
    for (int r = 0; r < 4; ++r)
        dwt[(size_t)(f0 + ty + 8 * r) * DD + d0 + tx] = t[tx][ty + 8 * r];
}

// ---------------- WMMA fragment loaders (per CDNA5 ISA 16-bit layouts) ----------------
// A 16x32: lanes 0-15 = rows, elems 0..7 = K(h*8..h*8+7), elems 8..15 = K(h*8+16..)
__device__ __forceinline__ v16bf frag_a(const bf16_t* base, int row, int lane)
{
    int r = lane & 15, h = lane >> 4;
    const bf16_t* p = base + (row + r) * LDSS + h * 8;
    v8bf lo = *(const v8bf*)p;
    v8bf hi = *(const v8bf*)(p + 16);
    return __builtin_shufflevector(lo, hi, 0,1,2,3,4,5,6,7,8,9,10,11,12,13,14,15);
}
// B 32x16 (enc_w row = one column of B): elems 0..7 = K(h*8..h*8+7), elems 8..15 = K(h*8+16..)
__device__ __forceinline__ v16bf frag_b(const bf16_t* base, int row, int lane)
{
    int r = lane & 15, h = lane >> 4;
    const bf16_t* p = base + (row + r) * LDSS + h * 8;
    v8bf lo = *(const v8bf*)p;
    v8bf hi = *(const v8bf*)(p + 16);
    return __builtin_shufflevector(lo, hi, 0,1,2,3,4,5,6,7,8,9,10,11,12,13,14,15);
}

// ---------------- encode GEMM: pre = Xc @ enc_w^T + enc_b ----------------
__global__ void __launch_bounds__(256)
k_encode_gemm(const bf16_t* __restrict__ xc,   // [BB][DD] bf16
              const bf16_t* __restrict__ ew,   // [FF][DD] bf16
              const float*  __restrict__ eb,   // [FF]
              float* __restrict__ pre)         // [BB][FF] (sparse region of d_out)
{
    __shared__ bf16_t Ash[2][128 * LDSS];
    __shared__ bf16_t Bsh[2][128 * LDSS];

    const int tid  = threadIdx.x;
    const int lane = tid & 31;
    const int wave = tid >> 5;
    const int wm   = wave & 1;    // 2 wave-rows -> 64 M each
    const int wn   = wave >> 1;   // 4 wave-cols -> 32 N each

    const int m0 = blockIdx.y * 128;
    const int n0 = blockIdx.x * 128;

    // each thread stages two 16B chunks per tile (128 rows x 32 k of bf16)
    const int r0 = tid >> 2,          s0 = (tid & 3) * 8;
    const int r1 = (tid + 256) >> 2,  s1 = ((tid + 256) & 3) * 8;

    v8f zero = {};
    v8f acc[4][2];
#pragma unroll
    for (int a = 0; a < 4; ++a)
#pragma unroll
        for (int b = 0; b < 2; ++b)
            acc[a][b] = zero;

    // prologue: stage K-tile 0 into buffer 0
    {
        const bf16_t* ag = xc + (size_t)m0 * DD;
        const bf16_t* bg = ew + (size_t)n0 * DD;
        cp16(ag + (size_t)r0 * DD + s0, Ash[0] + r0 * LDSS + s0);
        cp16(ag + (size_t)r1 * DD + s1, Ash[0] + r1 * LDSS + s1);
        cp16(bg + (size_t)r0 * DD + s0, Bsh[0] + r0 * LDSS + s0);
        cp16(bg + (size_t)r1 * DD + s1, Bsh[0] + r1 * LDSS + s1);
    }

    const int NT = DD / 32;
    for (int it = 0; it < NT; ++it) {
        const int cur = it & 1, nxt = cur ^ 1;

        if (it + 1 < NT) {  // stage next K-tile while computing this one
            const int kt = (it + 1) * 32;
            const bf16_t* ag = xc + (size_t)m0 * DD + kt;
            const bf16_t* bg = ew + (size_t)n0 * DD + kt;
            cp16(ag + (size_t)r0 * DD + s0, Ash[nxt] + r0 * LDSS + s0);
            cp16(ag + (size_t)r1 * DD + s1, Ash[nxt] + r1 * LDSS + s1);
            cp16(bg + (size_t)r0 * DD + s0, Bsh[nxt] + r0 * LDSS + s0);
            cp16(bg + (size_t)r1 * DD + s1, Bsh[nxt] + r1 * LDSS + s1);
            wait_async<4>();   // oldest 4 (current tile) complete
        } else {
            wait_async<0>();
        }
        __syncthreads();       // all waves' current-tile stages visible

        v16bf bfr[2];
#pragma unroll
        for (int ni = 0; ni < 2; ++ni)
            bfr[ni] = frag_b(Bsh[cur], wn * 32 + ni * 16, lane);
#pragma unroll
        for (int mi = 0; mi < 4; ++mi) {
            v16bf af = frag_a(Ash[cur], wm * 64 + mi * 16, lane);
#pragma unroll
            for (int ni = 0; ni < 2; ++ni)
                acc[mi][ni] = __builtin_amdgcn_wmma_f32_16x16x32_bf16(
                    false, af, false, bfr[ni], (short)0, acc[mi][ni], false, false);
        }
        __syncthreads();       // protect buffer reuse next iteration
    }

    // epilogue: C/D layout: VGPR i -> M = (lane>>4)*8 + i, N = lane&15
    const int nl = lane & 15;
    const int mh = (lane >> 4) * 8;
#pragma unroll
    for (int ni = 0; ni < 2; ++ni) {
        int col = n0 + wn * 32 + ni * 16 + nl;
        float bias = eb[col];
#pragma unroll
        for (int mi = 0; mi < 4; ++mi) {
            int rb = m0 + wm * 64 + mi * 16 + mh;
#pragma unroll
            for (int i = 0; i < 8; ++i)
                pre[(size_t)(rb + i) * FF + col] = acc[mi][ni][i] + bias;
        }
    }
}

// ---------------- top-k + scatter + decode ----------------
__device__ __forceinline__ unsigned mkey(float f)
{
    unsigned u = __float_as_uint(f);
    return (u & 0x80000000u) ? ~u : (u | 0x80000000u);  // monotonic float->uint
}

__global__ void __launch_bounds__(256)
k_topk_decode(float* __restrict__ pre,        // [BB][FF]: pre_act in, sparse out (in place)
              const float* __restrict__ dwt,  // [FF][DD] dec_w^T
              const float* __restrict__ db,   // [DD]
              float* __restrict__ recon)      // [BB][DD]
{
    __shared__ unsigned hist[256];
    __shared__ unsigned s_prefix;
    __shared__ int s_remK, s_n;
    __shared__ int   s_idx[64];
    __shared__ float s_val[64];

    const int row = blockIdx.x;
    const int tid = threadIdx.x;
    float* rowp = pre + (size_t)row * FF;

    float v[32];
#pragma unroll
    for (int j4 = 0; j4 < 8; ++j4) {
        float4 t = *(const float4*)(rowp + tid * 32 + j4 * 4);
        v[j4*4+0] = t.x; v[j4*4+1] = t.y; v[j4*4+2] = t.z; v[j4*4+3] = t.w;
    }
    unsigned kx[32];
#pragma unroll
    for (int j = 0; j < 32; ++j) kx[j] = mkey(v[j]);

    if (tid == 0) { s_prefix = 0; s_remK = KK; s_n = 0; }

    // radix-select the KK-th largest key, one byte per pass (MSB first)
    for (int p = 3; p >= 0; --p) {
        hist[tid] = 0;
        __syncthreads();
        unsigned pf = s_prefix;
#pragma unroll
        for (int j = 0; j < 32; ++j) {
            unsigned kk = kx[j];
            bool m = (p == 3) || ((kk >> ((p + 1) * 8)) == pf);
            if (m) atomicAdd(&hist[(kk >> (p * 8)) & 255u], 1u);
        }
        __syncthreads();
        if (tid == 0) {
            int rem = s_remK;
            unsigned cum = 0;
            int b = 255;
            for (; b > 0; --b) {
                if (cum + hist[b] >= (unsigned)rem) break;
                cum += hist[b];
            }
            s_remK   = rem - (int)cum;
            s_prefix = (s_prefix << 8) | (unsigned)b;
        }
        __syncthreads();
    }
    const unsigned thr = s_prefix;

    // sparsify row in place; collect positive selected features
#pragma unroll
    for (int j4 = 0; j4 < 8; ++j4) {
        float o[4];
#pragma unroll
        for (int e = 0; e < 4; ++e) {
            int j = j4 * 4 + e;
            float r = (kx[j] >= thr) ? fmaxf(v[j], 0.f) : 0.f;
            o[e] = r;
            if (r > 0.f) {
                int pos = atomicAdd(&s_n, 1);
                if (pos < 64) { s_idx[pos] = tid * 32 + j; s_val[pos] = r; }
            }
        }
        *(float4*)(rowp + tid * 32 + j4 * 4) = make_float4(o[0], o[1], o[2], o[3]);
    }
    __syncthreads();
    int n = s_n; if (n > 64) n = 64;

    // decode: recon[row, d] = dec_b[d] + sum_sel val * dec_wT[idx][d]
    const int d0 = tid * 4;
    float4 acc = *(const float4*)(db + d0);
    for (int s = 0; s < n; ++s) {
        float w = s_val[s];
        const float4 r = *(const float4*)(dwt + (size_t)s_idx[s] * DD + d0);
        acc.x += w * r.x; acc.y += w * r.y; acc.z += w * r.z; acc.w += w * r.w;
    }
    *(float4*)(recon + (size_t)row * DD + d0) = acc;
}

// ---------------- launch ----------------
extern "C" void kernel_launch(void* const* d_in, const int* in_sizes, int n_in,
                              void* d_out, int out_size, void* d_ws, size_t ws_size,
                              hipStream_t stream)
{
    const float* x    = (const float*)d_in[0];
    const float* encw = (const float*)d_in[1];
    const float* encb = (const float*)d_in[2];
    const float* decw = (const float*)d_in[3];
    const float* decb = (const float*)d_in[4];

    float* out    = (float*)d_out;
    float* recon  = out;                       // [BB][DD]
    float* sparse = out + (size_t)BB * DD;     // [BB][FF]

    char* ws = (char*)d_ws;
    bf16_t* xc  = (bf16_t*)ws;                                              // 32 MiB
    bf16_t* ewb = (bf16_t*)(ws + (size_t)BB * DD * 2);                      // 16 MiB
    float*  dwt = (float*)(ws + (size_t)BB * DD * 2 + (size_t)FF * DD * 2); // 32 MiB

    k_center_bf16<<<(BB * DD / 8) / 256, 256, 0, stream>>>(x, decb, xc);
    k_cvt_bf16  <<<(FF * DD / 8) / 256, 256, 0, stream>>>(encw, ewb);

    dim3 gt(FF / 32, DD / 32);
    k_transpose<<<gt, 256, 0, stream>>>(decw, dwt);

    dim3 g(FF / 128, BB / 128);
    k_encode_gemm<<<g, 256, 0, stream>>>(xc, ewb, encb, sparse);

    k_topk_decode<<<BB, 256, 0, stream>>>(sparse, dwt, decb, recon);
}
